// MOMVFCC_34617436406162
// MI455X (gfx1250) — matrix operand order repaired and checked by reading, not compiled, not run on hardware
//
#include <hip/hip_runtime.h>
#include <cstdint>
#include <cstddef>

// Problem constants (match reference)
#define NN    6144
#define DD0   512
#define DD1   256
#define DD2   128
#define KK    10
#define ALPHA_C 1.0f
#define BETA_C  1.0f

typedef __attribute__((ext_vector_type(16))) __bf16 v16bf;
typedef __attribute__((ext_vector_type(8)))  __bf16 v8bf;
typedef __attribute__((ext_vector_type(8)))  float  v8f;

// ---------------------------------------------------------------- helpers
static __device__ __forceinline__ unsigned short f2bf(float f) {
  union { float f; unsigned u; } v; v.f = f;
  unsigned r = v.u + 0x7FFFu + ((v.u >> 16) & 1u);   // round-to-nearest-even
  return (unsigned short)(r >> 16);
}

// ---------------------------------------------------------------- converts
__global__ void k_f32_to_bf16(const float* __restrict__ s,
                              unsigned short* __restrict__ d, int n) {
  int i = blockIdx.x * blockDim.x + threadIdx.x;
  if (i < n) d[i] = f2bf(s[i]);
}

// src [R,C] f32 row-major -> dst [C,R] bf16 row-major (i.e. srcᵀ)
__global__ void k_transpose_to_bf16(const float* __restrict__ s,
                                    unsigned short* __restrict__ d, int R, int C) {
  int i = blockIdx.x * blockDim.x + threadIdx.x;
  if (i < R * C) { int r = i / C, c = i - r * C; d[c * R + r] = f2bf(s[i]); }
}

// ---------------------------------------------------------------- dual matvec
// f1[i] = H[i,:]·v1 ; f2[i] = H[i,:]·v2 ; one wave (32 lanes) per row
__global__ void k_dual_matvec(const float* __restrict__ H,
                              const float* __restrict__ v1,
                              const float* __restrict__ v2,
                              float* __restrict__ f1, float* __restrict__ f2, int D) {
  int lane = threadIdx.x & 31;
  int wave = threadIdx.x >> 5;
  int row  = blockIdx.x * 8 + wave;
  const float* h = H + (size_t)row * D;
  float s1 = 0.f, s2 = 0.f;
  for (int d = lane; d < D; d += 32) { float x = h[d]; s1 += x * v1[d]; s2 += x * v2[d]; }
  for (int o = 16; o > 0; o >>= 1) { s1 += __shfl_down(s1, o); s2 += __shfl_down(s2, o); }
  if (lane == 0) { f1[row] = s1; f2[row] = s2; }
}

// ---------------------------------------------------------------- attention
// E_ij = (A_ij!=0) ? exp(sigmoid(A_ij*(f1_i+f2_j))) : 0  (bf16), rden_i = 1/rowsum (or 1)
__global__ void k_attention(const float* __restrict__ A,
                            const float* __restrict__ f1,
                            const float* __restrict__ f2,
                            unsigned short* __restrict__ E,
                            float* __restrict__ rden, int n) {
  __shared__ float red[256];
  int row = blockIdx.x;
  const float* a = A + (size_t)row * n;
  unsigned short* e = E + (size_t)row * n;
  float fi = f1[row];
  float s = 0.f;
  for (int j = threadIdx.x; j < n; j += 256) {
    float av = a[j];
    float ev = 0.f;
    if (av != 0.f) {
      float l  = av * (fi + f2[j]);
      float sg = 1.f / (1.f + __expf(-l));
      ev = __expf(sg);
    }
    e[j] = f2bf(ev);
    s += ev;
  }
  red[threadIdx.x] = s;
  __syncthreads();
  for (int st = 128; st > 0; st >>= 1) {
    if (threadIdx.x < st) red[threadIdx.x] += red[threadIdx.x + st];
    __syncthreads();
  }
  if (threadIdx.x == 0) { float d = red[0]; rden[row] = (d == 0.f) ? 1.f : 1.f / d; }
}

// ---------------------------------------------------------------- WMMA GEMM
// C[M,Nc] (f32) = rowscale ⊙ (A[M,K] (bf16) @ B[K,Nc] (bf16))
// block = 256 threads (8 waves); block tile = 256 rows x 64 cols; k-step 32.
// Wave w owns rows {w*16..+16} and {128+w*16..+16} (2 A frags, 8 accumulators):
// each staged B tile / B-fragment LDS read feeds 2 WMMAs.
// LDS B tile double-buffered -> single barrier per k-step, stage overlaps WMMA.
__global__ __launch_bounds__(256)
void k_wmma_gemm(const __bf16* __restrict__ Am, const __bf16* __restrict__ Bm,
                 float* __restrict__ Cm, const float* __restrict__ rscale,
                 int M, int K, int Nc) {
  __shared__ __align__(16) __bf16 Bt[2][64 * 32];   // [buf][col][k] transposed B tiles

  const int tid  = threadIdx.x;
  const int lane = tid & 31;
  const int wave = tid >> 5;
  const int cb   = blockIdx.x * 64;
  const int r0   = blockIdx.y * 256 + wave * 16;
  const int r1   = r0 + 128;

  // ISA 16-bit A layout: lanes0-15 -> M=lane, K in {0..7,16..23}; lanes16-31 -> K in {8..15,24..31}
  const int mrow = lane & 15;
  const int asel = (lane < 16) ? 0 : 8;
  const int koff = (lane < 16) ? 0 : 8;
  const __bf16* ap0 = Am + (size_t)(r0 + mrow) * K + asel;
  const __bf16* ap1 = Am + (size_t)(r1 + mrow) * K + asel;

  // B staging mapping: thread -> (k-within-tile, 8-col segment)
  const int skk  = tid >> 3;   // 0..31
  const int sseg = tid & 7;    // 0..7
  const __bf16* bgp = Bm + (size_t)skk * Nc + cb + sseg * 8;

  const v8f vzero = {0.f, 0.f, 0.f, 0.f, 0.f, 0.f, 0.f, 0.f};
  v8f acc[8];
#pragma unroll
  for (int c = 0; c < 8; ++c) acc[c] = vzero;

  // prologue: stage tile k0=0 into buffer 0
  {
    v8bf val = *(const v8bf*)bgp;
#pragma unroll
    for (int j = 0; j < 8; ++j) Bt[0][(sseg * 8 + j) * 32 + skk] = val[j];
  }
  __syncthreads();

  int buf = 0;
  for (int k0 = 0; k0 < K; k0 += 32) {
    // stage next tile into the other buffer (overlaps with this step's WMMAs)
    if (k0 + 32 < K) {
      v8bf val = *(const v8bf*)(bgp + (size_t)(k0 + 32) * Nc);
#pragma unroll
      for (int j = 0; j < 8; ++j) Bt[buf ^ 1][(sseg * 8 + j) * 32 + skk] = val[j];
      __builtin_prefetch(ap0 + k0 + 32, 0, 1);
      __builtin_prefetch(ap1 + k0 + 32, 0, 1);
    }

    // A fragments: two contiguous 8-half chunks per lane, per row group
    v8bf a0lo = *(const v8bf*)(ap0 + k0);
    v8bf a0hi = *(const v8bf*)(ap0 + k0 + 16);
    v8bf a1lo = *(const v8bf*)(ap1 + k0);
    v8bf a1hi = *(const v8bf*)(ap1 + k0 + 16);
    v16bf a0, a1;
#pragma unroll
    for (int i = 0; i < 8; ++i) {
      a0[i] = a0lo[i]; a0[i + 8] = a0hi[i];
      a1[i] = a1lo[i]; a1[i + 8] = a1hi[i];
    }

#pragma unroll
    for (int c = 0; c < 4; ++c) {
      const __bf16* bp = &Bt[buf][(c * 16 + mrow) * 32 + koff];
      v8bf blo = *(const v8bf*)bp;
      v8bf bhi = *(const v8bf*)(bp + 16);
      v16bf b;
#pragma unroll
      for (int i = 0; i < 8; ++i) { b[i] = blo[i]; b[i + 8] = bhi[i]; }
      acc[c] = __builtin_amdgcn_wmma_f32_16x16x32_bf16(
          false, a0, false, b, (short)0, acc[c], false, false);
      acc[4 + c] = __builtin_amdgcn_wmma_f32_16x16x32_bf16(
          false, a1, false, b, (short)0, acc[4 + c], false, false);
    }
    __syncthreads();   // Bt[buf^1] writes done; Bt[buf] reads done
    buf ^= 1;
  }

  // epilogue: C/D layout: VGPR r -> M = rbase + r + (lane<16?0:8), N = cb + c*16 + lane%16
#pragma unroll
  for (int g = 0; g < 2; ++g) {
    int rb = (g == 0) ? r0 : r1;
#pragma unroll
    for (int c = 0; c < 4; ++c) {
      int col = cb + c * 16 + mrow;
#pragma unroll
      for (int r = 0; r < 8; ++r) {
        int row = rb + r + ((lane < 16) ? 0 : 8);
        float sc = rscale ? rscale[row] : 1.0f;
        Cm[(size_t)row * Nc + col] = acc[g * 4 + c][r] * sc;
      }
    }
  }
}

// ---------------------------------------------------------------- fuse + q
// H_F = H + beta*H2 ; q = student-t soft assignment vs mu[K,128]
__global__ void k_fuse_q(const float* __restrict__ H, const float* __restrict__ H2,
                         float* __restrict__ HF, float* __restrict__ q,
                         const float* __restrict__ mu) {
  __shared__ float hf[DD2];
  __shared__ float qv[16];
  int row = blockIdx.x, t = threadIdx.x;   // 128 threads
  float v = H[(size_t)row * DD2 + t] + BETA_C * H2[(size_t)row * DD2 + t];
  HF[(size_t)row * DD2 + t] = v;
  hf[t] = v;
  __syncthreads();
  if (t < KK) {
    float d2 = 0.f;
    for (int d = 0; d < DD2; ++d) { float df = hf[d] - mu[t * DD2 + d]; d2 += df * df; }
    qv[t] = powf(1.0f / (1.0f + d2 / ALPHA_C), (ALPHA_C + 1.0f) * 0.5f);
  }
  __syncthreads();
  if (t < KK) {
    float ssum = 0.f;
    for (int k = 0; k < KK; ++k) ssum += qv[k];
    q[(size_t)row * KK + t] = qv[t] / ssum;
  }
}

// ---------------------------------------------------------------- host side
static inline void launch_gemm(const __bf16* Am, const __bf16* Bm, float* Cm,
                               const float* rs, int M, int K, int Nc, hipStream_t s) {
  dim3 g(Nc / 64, M / 256);
  k_wmma_gemm<<<g, 256, 0, s>>>(Am, Bm, Cm, rs, M, K, Nc);
}
static inline void cvt(const float* src, __bf16* dst, int n, hipStream_t s) {
  k_f32_to_bf16<<<(n + 255) / 256, 256, 0, s>>>(src, (unsigned short*)dst, n);
}

static void run_autoenc(const float* A, const float* X,
                        const float* W1, const float* v11, const float* v12,
                        const float* W2, const float* v21, const float* v22,
                        float* outH, float* outX,
                        __bf16* E1, __bf16* E2, float* rd1, float* rd2,
                        float* f1, float* f2,
                        __bf16* Xb, __bf16* W1b, __bf16* W1t, __bf16* W2b, __bf16* W2t,
                        float* P0, __bf16* P0b, __bf16* P1b, __bf16* P2b,
                        hipStream_t s) {
  const int N = NN;
  // weight / input conversions
  cvt(W1, W1b, DD0 * DD1, s);
  k_transpose_to_bf16<<<(DD0 * DD1 + 255) / 256, 256, 0, s>>>(W1, (unsigned short*)W1t, DD0, DD1);
  cvt(W2, W2b, DD1 * DD2, s);
  k_transpose_to_bf16<<<(DD1 * DD2 + 255) / 256, 256, 0, s>>>(W2, (unsigned short*)W2t, DD1, DD2);
  cvt(X, Xb, N * DD0, s);

  // ---- encoder layer 1: H1 = X @ W1  [N,D1]
  launch_gemm(Xb, W1b, P0, nullptr, N, DD0, DD1, s);
  cvt(P0, P0b, N * DD1, s);
  k_dual_matvec<<<N / 8, 256, 0, s>>>(P0, v11, v12, f1, f2, DD1);
  k_attention<<<N, 256, 0, s>>>(A, f1, f2, (unsigned short*)E1, rd1, N);
  launch_gemm(E1, P0b, P0, rd1, N, N, DD1, s);          // H = C1 @ H1
  cvt(P0, P1b, N * DD1, s);

  // ---- encoder layer 2: H2 = H @ W2  [N,D2]
  launch_gemm(P1b, W2b, P0, nullptr, N, DD1, DD2, s);
  cvt(P0, P0b, N * DD2, s);
  k_dual_matvec<<<N / 8, 256, 0, s>>>(P0, v21, v22, f1, f2, DD2);
  k_attention<<<N, 256, 0, s>>>(A, f1, f2, (unsigned short*)E2, rd2, N);
  launch_gemm(E2, P0b, outH, rd2, N, N, DD2, s);        // H_enc -> d_out
  cvt(outH, P2b, N * DD2, s);

  // ---- decoder (tied weights, cached attention)
  launch_gemm(P2b, W2t, P0, nullptr, N, DD2, DD1, s);   // H_enc @ W2ᵀ
  cvt(P0, P0b, N * DD1, s);
  launch_gemm(E2, P0b, P0, rd2, N, N, DD1, s);          // C2 @ ...
  cvt(P0, P1b, N * DD1, s);
  launch_gemm(P1b, W1t, P0, nullptr, N, DD1, DD0, s);   // ... @ W1ᵀ  [N,D0]
  cvt(P0, P0b, N * DD0, s);
  launch_gemm(E1, P0b, outX, rd1, N, N, DD0, s);        // X_rec -> d_out
}

extern "C" void kernel_launch(void* const* d_in, const int* in_sizes, int n_in,
                              void* d_out, int out_size, void* d_ws, size_t ws_size,
                              hipStream_t stream) {
  (void)in_sizes; (void)n_in; (void)out_size; (void)ws_size;
  const float* A    = (const float*)d_in[0];
  const float* A2   = (const float*)d_in[1];
  const float* X    = (const float*)d_in[2];
  const float* X2   = (const float*)d_in[3];
  const float* W11  = (const float*)d_in[4];
  const float* v111 = (const float*)d_in[5];
  const float* v112 = (const float*)d_in[6];
  const float* W12  = (const float*)d_in[7];
  const float* v121 = (const float*)d_in[8];
  const float* v122 = (const float*)d_in[9];
  const float* W21  = (const float*)d_in[10];
  const float* v211 = (const float*)d_in[11];
  const float* v212 = (const float*)d_in[12];
  const float* W22  = (const float*)d_in[13];
  const float* v221 = (const float*)d_in[14];
  const float* v222 = (const float*)d_in[15];
  const float* mu   = (const float*)d_in[16];

  float* out = (float*)d_out;
  const size_t N = NN;
  float* o_HF = out;                  // [N,128]
  float* o_q  = o_HF + N * DD2;       // [N,10]
  float* o_H  = o_q  + N * KK;        // [N,128]
  float* o_H2 = o_H  + N * DD2;       // [N,128]
  float* o_X  = o_H2 + N * DD2;       // [N,512]
  float* o_X2 = o_X  + N * DD0;       // [N,512]

  // ---- workspace carve-out (~190 MB)
  char* w = (char*)d_ws;
  auto carve = [&](size_t bytes) -> char* {
    char* p = w; w += (bytes + 255) & ~(size_t)255; return p;
  };
  __bf16* E1  = (__bf16*)carve(N * N * 2);
  __bf16* E2  = (__bf16*)carve(N * N * 2);
  float*  rd1 = (float*) carve(N * 4);
  float*  rd2 = (float*) carve(N * 4);
  float*  f1  = (float*) carve(N * 4);
  float*  f2  = (float*) carve(N * 4);
  __bf16* Xb  = (__bf16*)carve(N * DD0 * 2);
  __bf16* W1b = (__bf16*)carve(DD0 * DD1 * 2);
  __bf16* W1t = (__bf16*)carve(DD0 * DD1 * 2);
  __bf16* W2b = (__bf16*)carve(DD1 * DD2 * 2);
  __bf16* W2t = (__bf16*)carve(DD1 * DD2 * 2);
  float*  P0  = (float*) carve(N * DD0 * 4);
  __bf16* P0b = (__bf16*)carve(N * DD0 * 2);
  __bf16* P1b = (__bf16*)carve(N * DD0 * 2);
  __bf16* P2b = (__bf16*)carve(N * DD0 * 2);

  run_autoenc(A,  X,  W11, v111, v112, W12, v121, v122, o_H,  o_X,
              E1, E2, rd1, rd2, f1, f2, Xb, W1b, W1t, W2b, W2t,
              P0, P0b, P1b, P2b, stream);
  run_autoenc(A2, X2, W21, v211, v212, W22, v221, v222, o_H2, o_X2,
              E1, E2, rd1, rd2, f1, f2, Xb, W1b, W1t, W2b, W2t,
              P0, P0b, P1b, P2b, stream);

  k_fuse_q<<<NN, DD2, 0, stream>>>(o_H, o_H2, o_HF, o_q, mu);
}
